// PyramidAxialEncoder_6408091205812
// MI455X (gfx1250) — matrix operand order repaired and checked by reading, not compiled, not run on hardware
//
#include <hip/hip_runtime.h>
#include <hip/hip_bf16.h>
#include <math.h>

typedef __attribute__((ext_vector_type(16))) __bf16 v16bf;
typedef __attribute__((ext_vector_type(8)))  float  v8f;
typedef unsigned short u16;
typedef unsigned int   u32;

#define DEV __device__ __forceinline__

// ---------------- bf16 helpers ----------------
DEV u16 f2bf(float f) {
  u32 u = __float_as_uint(f);
  u32 r = ((u >> 16) & 1u) + 0x7fffu;   // round-to-nearest-even
  return (u16)((u + r) >> 16);
}

union ABf  { v16bf v; uint4 q[2]; u16 s[16]; };
union ACCf { v8f v; float f[8]; };

DEV v8f wmma_bf16(const ABf& a, const ABf& b, v8f c) {
  return __builtin_amdgcn_wmma_f32_16x16x32_bf16(false, a.v, false, b.v, (short)0, c,
                                                 false, false);
}

// block of 128 threads (4 waves, wave32): sum across the block
DEV float blocksum128(float v, float* red) {
  int t = threadIdx.x, wid = t >> 5, ln = t & 31;
  for (int o = 16; o; o >>= 1) v += __shfl_xor(v, o, 32);
  __syncthreads();
  if (ln == 0) red[wid] = v;
  __syncthreads();
  return red[0] + red[1] + red[2] + red[3];
}

// ---------------- weight conversion ----------------
// W[K][N] f32 -> WT[N][K] bf16
__global__ void k_wT(const float* __restrict__ W, u16* __restrict__ WT, int K, int N) {
  int i = blockIdx.x * 256 + threadIdx.x;
  if (i >= K * N) return;
  int k = i / N, n = i - k * N;
  WT[(size_t)n * K + k] = f2bf(W[i]);
}
// plain convert
__global__ void k_wC(const float* __restrict__ W, u16* __restrict__ D, int n) {
  int i = blockIdx.x * 256 + threadIdx.x;
  if (i < n) D[i] = f2bf(W[i]);
}
__global__ void k_cvt(const float* __restrict__ S, u16* __restrict__ D, int n) {
  int i = blockIdx.x * 256 + threadIdx.x;
  if (i < n) D[i] = f2bf(S[i]);
}

// ---------------- geometry ----------------
// c_embed[cam][ch] = cam_w[ch] . E_inv[cam][:,3]
__global__ __launch_bounds__(128)
void k_cam(const float* __restrict__ E_inv, const float* __restrict__ cam_w,
           float* __restrict__ c_embed) {
  int cam = blockIdx.x, ch = threadIdx.x;
  const float* E = E_inv + cam * 16;
  float c0 = E[3], c1 = E[7], c2 = E[11], c3 = E[15];
  const float* w = cam_w + ch * 4;
  c_embed[cam * 128 + ch] = w[0] * c0 + w[1] * c1 + w[2] * c2 + w[3] * c3;
}

// per (cam, pixel of 32x64): img_embed (l2norm) + BN/ReLU channels-last bf16
__global__ __launch_bounds__(128)
void k_img(const float* __restrict__ I_inv, const float* __restrict__ E_inv,
           const float* __restrict__ img_w, const float* __restrict__ c_embed,
           const float* __restrict__ feat,
           const float* __restrict__ fp_g, const float* __restrict__ fp_b,
           const float* __restrict__ fl_g, const float* __restrict__ fl_b,
           float* __restrict__ img_embed, u16* __restrict__ ffr_fp,
           u16* __restrict__ ffr_fl) {
  __shared__ float red[4];
  int bp = blockIdx.x;
  int cam = bp >> 11, p = bp & 2047;
  int py = p >> 6, px = p & 63;
  float gx = (float)px * (512.0f / 63.0f);
  float gy = (float)py * (256.0f / 31.0f);
  const float* Ii = I_inv + cam * 9;
  float c3v[3];
#pragma unroll
  for (int i = 0; i < 3; i++) c3v[i] = Ii[i * 3] * gx + Ii[i * 3 + 1] * gy + Ii[i * 3 + 2];
  const float* Ei = E_inv + cam * 16;
  float d4[4];
#pragma unroll
  for (int i = 0; i < 4; i++)
    d4[i] = Ei[i * 4] * c3v[0] + Ei[i * 4 + 1] * c3v[1] + Ei[i * 4 + 2] * c3v[2] + Ei[i * 4 + 3];
  int ch = threadIdx.x;
  const float* w = img_w + ch * 4;
  float de = w[0] * d4[0] + w[1] * d4[1] + w[2] * d4[2] + w[3] * d4[3]
           - c_embed[cam * 128 + ch];
  float ss = blocksum128(de * de, red);
  float inv = 1.0f / fmaxf(sqrtf(ss), 1e-12f);
  img_embed[(size_t)bp * 128 + ch] = de * inv;
  // BN(1x1)+ReLU, channels-last bf16
  float f = feat[((size_t)cam * 128 + ch) * 2048 + p];
  const float bns = 0.9999950000374997f; // 1/sqrt(1+1e-5)
  float vfp = fmaxf(f * (fp_g[ch] * bns) + fp_b[ch], 0.0f);
  float vfl = fmaxf(f * (fl_g[ch] * bns) + fl_b[ch], 0.0f);
  ffr_fp[(size_t)bp * 128 + ch] = f2bf(vfp);
  ffr_fl[(size_t)bp * 128 + ch] = f2bf(vfl);
}

// per (cam, pixel of 64x64): query = l2norm(w_embed - c_embed) + x  (f32, channels-last)
__global__ __launch_bounds__(128)
void k_query(const float* __restrict__ bev_w, const float* __restrict__ bev_b,
             const float* __restrict__ bevg, const float* __restrict__ c_embed,
             const float* __restrict__ x, float* __restrict__ query) {
  __shared__ float red[4];
  int bp = blockIdx.x;
  int cam = bp >> 12, p = bp & 4095;
  int ch = threadIdx.x;
  float w0 = bevg[p], w1 = bevg[4096 + p];
  float we = bev_w[ch * 2] * w0 + bev_w[ch * 2 + 1] * w1 + bev_b[ch];
  float de = we - c_embed[cam * 128 + ch];
  float ss = blocksum128(de * de, red);
  float inv = 1.0f / fmaxf(sqrtf(ss), 1e-12f);
  query[(size_t)bp * 128 + ch] = de * inv + x[(size_t)ch * 4096 + p];
}

// ---------------- LayerNorm (rows of 128) -> bf16 ----------------
__global__ __launch_bounds__(128)
void k_ln(const float* __restrict__ X, const float* __restrict__ g,
          const float* __restrict__ b, u16* __restrict__ Y) {
  __shared__ float red[4];
  int r = blockIdx.x, t = threadIdx.x;
  float xv = X[(size_t)r * 128 + t];
  float mean = blocksum128(xv, red) * (1.0f / 128.0f);
  float d = xv - mean;
  float var = blocksum128(d * d, red) * (1.0f / 128.0f);
  float y = d * rsqrtf(var + 1e-5f) * g[t] + b[t];
  Y[(size_t)r * 128 + t] = f2bf(y);
}

// final post-LN + transpose to (128,64,64)
__global__ __launch_bounds__(128)
void k_postln(const float* __restrict__ X, const float* __restrict__ g,
              const float* __restrict__ b, float* __restrict__ out) {
  __shared__ float red[4];
  int p = blockIdx.x, t = threadIdx.x;
  float xv = X[(size_t)p * 128 + t];
  float mean = blocksum128(xv, red) * (1.0f / 128.0f);
  float d = xv - mean;
  float var = blocksum128(d * d, red) * (1.0f / 128.0f);
  out[(size_t)t * 4096 + p] = d * rsqrtf(var + 1e-5f) * g[t] + b[t];
}

// mean over 6 cameras + bf16 convert: Oh[6*4096][128] -> out[4096][128]
__global__ void k_mean6(const float* __restrict__ Oh, u16* __restrict__ out) {
  int i = blockIdx.x * 256 + threadIdx.x;
  if (i >= 4096 * 128) return;
  int p = i >> 7, c = i & 127;
  float s = 0.0f;
#pragma unroll
  for (int cam = 0; cam < 6; cam++) s += Oh[((size_t)cam * 4096 + p) * 128 + c];
  out[i] = f2bf(s * (1.0f / 6.0f));
}

// ---------------- generic WMMA GEMM (32x32 wave tile: 4 WMMA per 4 frag loads) ----
// C[R][N] = (A[R][K](bf16) @ WT[N][K]^T + bias) * alpha  (+addA[r*N+n]) (+addT[n*R+r]) (gelu)
__global__ __launch_bounds__(256)
void k_gemm(const u16* __restrict__ A, const u16* __restrict__ BT,
            const float* __restrict__ bias, const float* __restrict__ addA,
            const float* __restrict__ addT, float* __restrict__ Cf,
            u16* __restrict__ Cb, int R, int N, int K, float alpha, int gelu) {
  int wave = threadIdx.x >> 5, lane = threadIdx.x & 31;
  int half = lane >> 4, lid = lane & 15;
  int tile = blockIdx.x * 8 + wave;
  int tnc = N >> 5;
  int tm = tile / tnc, tn = tile - tm * tnc;
  if (tm * 32 >= R) return;
  int m0 = tm * 32, n0 = tn * 32;
  ACCf acc00, acc01, acc10, acc11;
#pragma unroll
  for (int g = 0; g < 8; g++) {
    acc00.f[g] = 0.0f; acc01.f[g] = 0.0f; acc10.f[g] = 0.0f; acc11.f[g] = 0.0f;
  }
  const u16* arow0 = A + (size_t)(m0 + lid) * K;
  const u16* arow1 = A + (size_t)(m0 + 16 + lid) * K;
  const u16* brow0 = BT + (size_t)(n0 + lid) * K;
  const u16* brow1 = BT + (size_t)(n0 + 16 + lid) * K;
  for (int k0 = 0; k0 < K; k0 += 32) {
    if (k0 + 32 < K) {  // speculative prefetch of next K-block (global_prefetch_b8)
      __builtin_prefetch(arow0 + k0 + 32, 0, 1);
      __builtin_prefetch(brow0 + k0 + 32, 0, 1);
    }
    ABf a0, a1, b0, b1;
    a0.q[0] = *(const uint4*)(arow0 + k0 + half * 8);
    a0.q[1] = *(const uint4*)(arow0 + k0 + 16 + half * 8);
    a1.q[0] = *(const uint4*)(arow1 + k0 + half * 8);
    a1.q[1] = *(const uint4*)(arow1 + k0 + 16 + half * 8);
    b0.q[0] = *(const uint4*)(brow0 + k0 + half * 8);
    b0.q[1] = *(const uint4*)(brow0 + k0 + 16 + half * 8);
    b1.q[0] = *(const uint4*)(brow1 + k0 + half * 8);
    b1.q[1] = *(const uint4*)(brow1 + k0 + 16 + half * 8);
    acc00.v = wmma_bf16(a0, b0, acc00.v);
    acc01.v = wmma_bf16(a0, b1, acc01.v);
    acc10.v = wmma_bf16(a1, b0, acc10.v);
    acc11.v = wmma_bf16(a1, b1, acc11.v);
  }
#pragma unroll
  for (int mi = 0; mi < 2; mi++) {
#pragma unroll
    for (int nj = 0; nj < 2; nj++) {
      const ACCf& ac = mi == 0 ? (nj == 0 ? acc00 : acc01) : (nj == 0 ? acc10 : acc11);
      int col = n0 + nj * 16 + lid;
#pragma unroll
      for (int g = 0; g < 8; g++) {
        int row = m0 + mi * 16 + g + (half ? 8 : 0);
        float v = ac.f[g];
        if (bias) v += bias[col];
        v *= alpha;
        if (addA) v += addA[(size_t)row * N + col];
        if (addT) v += addT[(size_t)col * R + row];
        if (gelu) v = 0.5f * v * (1.0f + erff(v * 0.70710678118f));
        if (Cf) Cf[(size_t)row * N + col] = v;
        if (Cb) Cb[(size_t)row * N + col] = f2bf(v);
      }
    }
  }
}

// ---------------- windowed cross attention (flash-style) ----------------
// mode 1: q rows = 6*256 per window (window gather), out rows include camera
// mode 2: q rows = 256 per window (camera-broadcast collapsed), grid-window keys
__global__ __launch_bounds__(256)
void k_attn(const u16* __restrict__ Qh, const u16* __restrict__ Kh,
            const u16* __restrict__ Vh, float* __restrict__ Oh, int mode) {
  __shared__ __align__(16) u16 sV[8][32][32]; // [wave][col-in-head][k-in-block]
  __shared__ __align__(16) u16 sP[8][16][32]; // [wave][q-row][k-in-block]
  int wave = threadIdx.x >> 5, lane = threadIdx.x & 31;
  int half = lane >> 4, lid = lane & 15;
  int nblk = (mode == 1) ? 12 : 2;
  int bw = blockIdx.x;
  int win = bw / (4 * nblk);
  int rem = bw - win * (4 * nblk);
  int head = rem / nblk;
  int qb = rem - head * nblk;
  int X = win >> 2, Y = win & 3;
  int m0 = (qb * 8 + wave) * 16;

  auto qmap = [&](int i) -> size_t {
    if (mode == 1) {
      int cam = i >> 8, w = i & 255;
      int wy = w >> 4, wx = w & 15;
      return (size_t)cam * 4096 + (size_t)((X * 16 + wy) * 64 + (Y * 16 + wx));
    } else {
      int wy = i >> 4, wx = i & 15;
      return (size_t)((X * 16 + wy) * 64 + (Y * 16 + wx));
    }
  };
  auto kmap = [&](int j) -> size_t {
    int cam = j >> 7, w = j & 127;
    int ky = w >> 4, kx = w & 15;
    int prow = (mode == 1) ? ((X * 8 + ky) * 64 + (Y * 16 + kx))
                           : ((ky * 4 + X) * 64 + (kx * 4 + Y));
    return (size_t)cam * 2048 + prow;
  };

  // Q A-fragment (loaded once; K-dim of the matmul = DHEAD = 32)
  ABf qa;
  {
    const u16* qp = Qh + qmap(m0 + lid) * 128 + head * 32;
    qa.q[0] = *(const uint4*)(qp + half * 8);
    qa.q[1] = *(const uint4*)(qp + 16 + half * 8);
  }

  ACCf o0, o1;
  float mrow[8], lrow[8];
#pragma unroll
  for (int g = 0; g < 8; g++) { o0.f[g] = 0.0f; o1.f[g] = 0.0f; mrow[g] = -1e30f; lrow[g] = 0.0f; }

  for (int k0 = 0; k0 < 768; k0 += 32) {
    // stage V rows (transposed) in LDS: sV[c][k]
    {
      const u16* vp = Vh + kmap(k0 + lane) * 128 + head * 32;
#pragma unroll
      for (int c = 0; c < 32; c++) sV[wave][c][lane] = vp[c];
    }
    // scores: two 16-key tiles
    ACCf s0, s1;
    {
      ABf kb;
      const u16* kp0 = Kh + kmap(k0 + lid) * 128 + head * 32 + half * 8;
      kb.q[0] = *(const uint4*)(kp0);
      kb.q[1] = *(const uint4*)(kp0 + 16);
      v8f z = {0.f,0.f,0.f,0.f,0.f,0.f,0.f,0.f};
      s0.v = wmma_bf16(qa, kb, z);
      const u16* kp1 = Kh + kmap(k0 + 16 + lid) * 128 + head * 32 + half * 8;
      kb.q[0] = *(const uint4*)(kp1);
      kb.q[1] = *(const uint4*)(kp1 + 16);
      s1.v = wmma_bf16(qa, kb, z);
    }
    // online softmax (row reductions over 16-lane halves)
#pragma unroll
    for (int g = 0; g < 8; g++) {
      float a0 = s0.f[g], a1 = s1.f[g];
      float mx = fmaxf(a0, a1);
      for (int o = 8; o; o >>= 1) mx = fmaxf(mx, __shfl_xor(mx, o, 32));
      float mn = fmaxf(mrow[g], mx);
      float sc = __expf(mrow[g] - mn);
      float p0 = __expf(a0 - mn), p1 = __expf(a1 - mn);
      float ps = p0 + p1;
      for (int o = 8; o; o >>= 1) ps += __shfl_xor(ps, o, 32);
      lrow[g] = lrow[g] * sc + ps;
      mrow[g] = mn;
      o0.f[g] *= sc; o1.f[g] *= sc;
      int row = g + (half ? 8 : 0);
      sP[wave][row][lid] = f2bf(p0);
      sP[wave][row][16 + lid] = f2bf(p1);
    }
    __syncthreads();
    // P (16x32) @ V (32x32): A from sP, B from sV (contiguous b128 LDS reads)
    ABf pa, vb;
    pa.q[0] = *(const uint4*)(&sP[wave][lid][half * 8]);
    pa.q[1] = *(const uint4*)(&sP[wave][lid][16 + half * 8]);
    vb.q[0] = *(const uint4*)(&sV[wave][lid][half * 8]);
    vb.q[1] = *(const uint4*)(&sV[wave][lid][16 + half * 8]);
    o0.v = wmma_bf16(pa, vb, o0.v);
    vb.q[0] = *(const uint4*)(&sV[wave][16 + lid][half * 8]);
    vb.q[1] = *(const uint4*)(&sV[wave][16 + lid][16 + half * 8]);
    o1.v = wmma_bf16(pa, vb, o1.v);
    __syncthreads();
  }
  // normalize & store
#pragma unroll
  for (int g = 0; g < 8; g++) {
    int i = m0 + g + (half ? 8 : 0);
    size_t orow = qmap(i);
    float inv = 1.0f / lrow[g];
    Oh[orow * 128 + head * 32 + lid] = o0.f[g] * inv;
    Oh[orow * 128 + head * 32 + 16 + lid] = o1.f[g] * inv;
  }
}

// ---------------- host ----------------
extern "C" void kernel_launch(void* const* d_in, const int* in_sizes, int n_in,
                              void* d_out, int out_size, void* d_ws, size_t ws_size,
                              hipStream_t stream) {
  (void)in_sizes; (void)n_in; (void)out_size; (void)ws_size;
  const float* x      = (const float*)d_in[0];
  const float* feat   = (const float*)d_in[1];
  const float* I_inv  = (const float*)d_in[2];
  const float* E_inv  = (const float*)d_in[3];
  const float* bevg   = (const float*)d_in[4];
  const float* fl_g   = (const float*)d_in[5];
  const float* fl_b   = (const float*)d_in[6];
  const float* fl_w   = (const float*)d_in[7];
  const float* fp_g   = (const float*)d_in[8];
  const float* fp_b   = (const float*)d_in[9];
  const float* fp_w   = (const float*)d_in[10];
  const float* bev_w  = (const float*)d_in[11];
  const float* bev_b  = (const float*)d_in[12];
  const float* img_w  = (const float*)d_in[13];
  const float* cam_w  = (const float*)d_in[14];
  struct Attn { const float *lnq_g,*lnq_b,*lnk_g,*lnk_b,*lnv_g,*lnv_b,*wq,*bq,*wk,*bk,*wv,*bv,*wo,*bo; };
  auto getA = [&](int base) {
    Attn a;
    a.lnq_g=(const float*)d_in[base+0]; a.lnq_b=(const float*)d_in[base+1];
    a.lnk_g=(const float*)d_in[base+2]; a.lnk_b=(const float*)d_in[base+3];
    a.lnv_g=(const float*)d_in[base+4]; a.lnv_b=(const float*)d_in[base+5];
    a.wq=(const float*)d_in[base+6];  a.bq=(const float*)d_in[base+7];
    a.wk=(const float*)d_in[base+8];  a.bk=(const float*)d_in[base+9];
    a.wv=(const float*)d_in[base+10]; a.bv=(const float*)d_in[base+11];
    a.wo=(const float*)d_in[base+12]; a.bo=(const float*)d_in[base+13];
    return a;
  };
  Attn A1 = getA(15), A2 = getA(29);
  const float* pn1_g = (const float*)d_in[43]; const float* pn1_b = (const float*)d_in[44];
  const float* m1_w1 = (const float*)d_in[45]; const float* m1_b1 = (const float*)d_in[46];
  const float* m1_w2 = (const float*)d_in[47]; const float* m1_b2 = (const float*)d_in[48];
  const float* pn2_g = (const float*)d_in[49]; const float* pn2_b = (const float*)d_in[50];
  const float* m2_w1 = (const float*)d_in[51]; const float* m2_b1 = (const float*)d_in[52];
  const float* m2_w2 = (const float*)d_in[53]; const float* m2_b2 = (const float*)d_in[54];
  const float* post_g = (const float*)d_in[55]; const float* post_b = (const float*)d_in[56];

  char* base = (char*)d_ws;
  size_t off = 0;
  auto alloc = [&](size_t bytes) -> void* {
    off = (off + 255) & ~(size_t)255;
    void* p = base + off;
    off += bytes;
    return p;
  };
  const size_t RK = 6 * 2048, RQ = 6 * 4096, P = 4096;

  float* c_embed   = (float*)alloc(6 * 128 * 4);
  float* img_embed = (float*)alloc(RK * 128 * 4);
  u16*   ffr_fp    = (u16*)alloc(RK * 128 * 2);
  u16*   ffr_fl    = (u16*)alloc(RK * 128 * 2);
  float* key_f32   = (float*)alloc(RK * 128 * 4);
  float* val_f32   = (float*)alloc(RK * 128 * 4);
  float* query_f32 = (float*)alloc(RQ * 128 * 4);
  u16*   Aq1       = (u16*)alloc(RQ * 128 * 2);
  u16*   Akey1     = (u16*)alloc(RK * 128 * 2);
  u16*   Aval1     = (u16*)alloc(RK * 128 * 2);
  u16*   Akey2     = (u16*)alloc(RK * 128 * 2);
  u16*   Aval2     = (u16*)alloc(RK * 128 * 2);
  u16*   Qh1       = (u16*)alloc(RQ * 128 * 2);
  u16*   Kh1       = (u16*)alloc(RK * 128 * 2);
  u16*   Vh1       = (u16*)alloc(RK * 128 * 2);
  u16*   Qh2       = (u16*)alloc(P * 128 * 2);
  u16*   Kh2       = (u16*)alloc(RK * 128 * 2);
  u16*   Vh2       = (u16*)alloc(RK * 128 * 2);
  float* Oh1       = (float*)alloc(RQ * 128 * 4);
  float* Oh2       = (float*)alloc(P * 128 * 4);
  u16*   Ohm1b     = (u16*)alloc(P * 128 * 2);
  u16*   Oh2b      = (u16*)alloc(P * 128 * 2);
  float* q1        = (float*)alloc(P * 128 * 4);
  float* q2        = (float*)alloc(P * 128 * 4);
  u16*   Aln       = (u16*)alloc(P * 128 * 2);
  u16*   Aq2       = (u16*)alloc(P * 128 * 2);
  u16*   hbuf      = (u16*)alloc(P * 256 * 2);
  u16* wq1T = (u16*)alloc(16384 * 2); u16* wk1T = (u16*)alloc(16384 * 2);
  u16* wv1T = (u16*)alloc(16384 * 2); u16* wo1T = (u16*)alloc(16384 * 2);
  u16* wq2T = (u16*)alloc(16384 * 2); u16* wk2T = (u16*)alloc(16384 * 2);
  u16* wv2T = (u16*)alloc(16384 * 2); u16* wo2T = (u16*)alloc(16384 * 2);
  u16* m1w1T = (u16*)alloc(32768 * 2); u16* m1w2T = (u16*)alloc(32768 * 2);
  u16* m2w1T = (u16*)alloc(32768 * 2); u16* m2w2T = (u16*)alloc(32768 * 2);
  u16* fpwB = (u16*)alloc(16384 * 2);  u16* flwB = (u16*)alloc(16384 * 2);

  auto gemm = [&](const u16* A, const u16* BT, const float* bias, const float* addA,
                  const float* addT, float* Cf, u16* Cb, int R, int N, int K,
                  float alpha, int gelu) {
    int tiles = (R / 32) * (N / 32);
    int blocks = (tiles + 7) / 8;
    k_gemm<<<blocks, 256, 0, stream>>>(A, BT, bias, addA, addT, Cf, Cb, R, N, K, alpha, gelu);
  };
  const float iscale = 0.17677669529663687f; // 1/sqrt(32)

  // weight conversions
  k_wT<<<64, 256, 0, stream>>>(A1.wq, wq1T, 128, 128);
  k_wT<<<64, 256, 0, stream>>>(A1.wk, wk1T, 128, 128);
  k_wT<<<64, 256, 0, stream>>>(A1.wv, wv1T, 128, 128);
  k_wT<<<64, 256, 0, stream>>>(A1.wo, wo1T, 128, 128);
  k_wT<<<64, 256, 0, stream>>>(A2.wq, wq2T, 128, 128);
  k_wT<<<64, 256, 0, stream>>>(A2.wk, wk2T, 128, 128);
  k_wT<<<64, 256, 0, stream>>>(A2.wv, wv2T, 128, 128);
  k_wT<<<64, 256, 0, stream>>>(A2.wo, wo2T, 128, 128);
  k_wT<<<128, 256, 0, stream>>>(m1_w1, m1w1T, 128, 256);
  k_wT<<<128, 256, 0, stream>>>(m1_w2, m1w2T, 256, 128);
  k_wT<<<128, 256, 0, stream>>>(m2_w1, m2w1T, 128, 256);
  k_wT<<<128, 256, 0, stream>>>(m2_w2, m2w2T, 256, 128);
  k_wC<<<64, 256, 0, stream>>>(fp_w, fpwB, 16384);
  k_wC<<<64, 256, 0, stream>>>(fl_w, flwB, 16384);

  // geometry + embeddings
  k_cam<<<6, 128, 0, stream>>>(E_inv, cam_w, c_embed);
  k_img<<<(int)RK, 128, 0, stream>>>(I_inv, E_inv, img_w, c_embed, feat,
                                     fp_g, fp_b, fl_g, fl_b, img_embed, ffr_fp, ffr_fl);
  k_query<<<(int)RQ, 128, 0, stream>>>(bev_w, bev_b, bevg, c_embed, x, query_f32);

  // key/val 1x1 convs (key += img_embed)
  gemm(ffr_fp, fpwB, nullptr, img_embed, nullptr, key_f32, nullptr, RK, 128, 128, 1.0f, 0);
  gemm(ffr_fl, flwB, nullptr, nullptr, nullptr, val_f32, nullptr, RK, 128, 128, 1.0f, 0);

  // LayerNorms for projections (both attention passes share key/val inputs)
  k_ln<<<(int)RQ, 128, 0, stream>>>(query_f32, A1.lnq_g, A1.lnq_b, Aq1);
  k_ln<<<(int)RK, 128, 0, stream>>>(key_f32, A1.lnk_g, A1.lnk_b, Akey1);
  k_ln<<<(int)RK, 128, 0, stream>>>(val_f32, A1.lnv_g, A1.lnv_b, Aval1);
  k_ln<<<(int)RK, 128, 0, stream>>>(key_f32, A2.lnk_g, A2.lnk_b, Akey2);
  k_ln<<<(int)RK, 128, 0, stream>>>(val_f32, A2.lnv_g, A2.lnv_b, Aval2);

  // attn1 projections (scale folded into Q)
  gemm(Aq1, wq1T, A1.bq, nullptr, nullptr, nullptr, Qh1, RQ, 128, 128, iscale, 0);
  gemm(Akey1, wk1T, A1.bk, nullptr, nullptr, nullptr, Kh1, RK, 128, 128, 1.0f, 0);
  gemm(Aval1, wv1T, A1.bv, nullptr, nullptr, nullptr, Vh1, RK, 128, 128, 1.0f, 0);

  // attention pass 1
  k_attn<<<16 * 4 * 12, 256, 0, stream>>>(Qh1, Kh1, Vh1, Oh1, 1);

  // mean over cameras (linear -> before wo), then wo + bo + skip(x)
  k_mean6<<<2048, 256, 0, stream>>>(Oh1, Ohm1b);
  gemm(Ohm1b, wo1T, A1.bo, nullptr, x, q1, nullptr, P, 128, 128, 1.0f, 0);

  // MLP1 (residual)
  k_ln<<<(int)P, 128, 0, stream>>>(q1, pn1_g, pn1_b, Aln);
  gemm(Aln, m1w1T, m1_b1, nullptr, nullptr, nullptr, hbuf, P, 256, 128, 1.0f, 1);
  gemm(hbuf, m1w2T, m1_b2, q1, nullptr, q1, nullptr, P, 128, 256, 1.0f, 0);

  // attn2 projections
  k_ln<<<(int)P, 128, 0, stream>>>(q1, A2.lnq_g, A2.lnq_b, Aq2);
  gemm(Aq2, wq2T, A2.bq, nullptr, nullptr, nullptr, Qh2, P, 128, 128, iscale, 0);
  gemm(Akey2, wk2T, A2.bk, nullptr, nullptr, nullptr, Kh2, RK, 128, 128, 1.0f, 0);
  gemm(Aval2, wv2T, A2.bv, nullptr, nullptr, nullptr, Vh2, RK, 128, 128, 1.0f, 0);

  // attention pass 2 (grid windows; camera broadcast collapsed => mean is identity)
  k_attn<<<16 * 4 * 2, 256, 0, stream>>>(Qh2, Kh2, Vh2, Oh2, 2);

  // wo2 + bo2 + skip(q1)
  k_cvt<<<2048, 256, 0, stream>>>(Oh2, Oh2b, (int)(P * 128));
  gemm(Oh2b, wo2T, A2.bo, q1, nullptr, q2, nullptr, P, 128, 128, 1.0f, 0);

  // MLP2 (residual)
  k_ln<<<(int)P, 128, 0, stream>>>(q2, pn2_g, pn2_b, Aln);
  gemm(Aln, m2w1T, m2_b1, nullptr, nullptr, nullptr, hbuf, P, 256, 128, 1.0f, 1);
  gemm(hbuf, m2w2T, m2_b2, q2, nullptr, q2, nullptr, P, 128, 256, 1.0f, 0);

  // post-LN + transpose to (1,128,64,64)
  k_postln<<<(int)P, 128, 0, stream>>>(q2, post_g, post_b, (float*)d_out);
}
